// GRU_24687472017935
// MI455X (gfx1250) — matrix operand order repaired and checked by reading, not compiled, not run on hardware
//
#include <hip/hip_runtime.h>
#include <hip/hip_bf16.h>
#include <math.h>

// ---------------------------------------------------------------------------
// GRU  (B=64, T=2048, I=256, H=512), gate order (reset, update, new)
//
// Phase 1: IG[t][b][n] = xs[b][t][:] @ w_ih[n][:] + b[n]    (bf16 WMMA GEMM)
// Phase 2: 2048 sequential steps, 16 cooperating workgroups:
//          hg = h @ w_hh^T  (w_hh slice TDM-staged to LDS, bf16 WMMA)
//          r=sig(ir+hr); z=sig(iz+hz); n=tanh(in+r*(hn+b_n)); h=n+z*(h-n)
// ---------------------------------------------------------------------------

#define B_  64
#define T_  2048
#define I_  256
#define H_  512
#define G3_ 1536   // 3*H

typedef __attribute__((ext_vector_type(16))) __bf16 v16bf;
typedef __attribute__((ext_vector_type(8)))  __bf16 bf16x8;
typedef __attribute__((ext_vector_type(8)))  float  v8f;
typedef __attribute__((ext_vector_type(4)))  float  f32x4;
typedef __attribute__((ext_vector_type(4)))  unsigned int u32x4;
typedef __attribute__((ext_vector_type(4)))  int    i32x4;
typedef __attribute__((ext_vector_type(8)))  int    i32x8;

// workspace layout (bytes, 256-aligned)
static constexpr size_t OFF_IG  = 0;                            // [T][B][3H] f32
static constexpr size_t SZ_IG   = (size_t)T_ * B_ * G3_ * 4;    // 768 MB
static constexpr size_t OFF_WIH = OFF_IG + SZ_IG;               // [3H][I] bf16
static constexpr size_t SZ_WIH  = (size_t)G3_ * I_ * 2;
static constexpr size_t OFF_WHH = OFF_WIH + SZ_WIH;             // [3H][H] bf16
static constexpr size_t SZ_WHH  = (size_t)G3_ * H_ * 2;
static constexpr size_t OFF_HF  = OFF_WHH + SZ_WHH;             // [2][B][H] f32 (ping-pong)
static constexpr size_t SZ_HF   = (size_t)2 * B_ * H_ * 4;
static constexpr size_t OFF_HB  = OFF_HF + SZ_HF;               // [2][B][H] bf16
static constexpr size_t SZ_HB   = (size_t)2 * B_ * H_ * 2;
static constexpr size_t OFF_BAR = OFF_HB + SZ_HB;               // int barrier

// ---- WMMA fragment loaders (CDNA5 16x16x32 bf16 layouts, CDNA5 ISA) -------

// A (16x32, MxK) from f32 row-major source; lane m=lane&15 holds
// K = {8g..8g+7} in halfwords 0..7 and K = {8g+16..8g+23} in halfwords 8..15.
__device__ __forceinline__ v16bf load_a_f32(const float* base, size_t lda,
                                            int k0, int lane) {
  const int m = lane & 15, g2 = lane >> 4;
  const float* p = base + (size_t)m * lda + k0 + 8 * g2;
  f32x4 a0 = *(const f32x4*)(p);
  f32x4 a1 = *(const f32x4*)(p + 4);
  f32x4 b0 = *(const f32x4*)(p + 16);
  f32x4 b1 = *(const f32x4*)(p + 20);
  v16bf r;
#pragma unroll
  for (int i = 0; i < 4; ++i) {
    r[i]      = (__bf16)a0[i];
    r[4 + i]  = (__bf16)a1[i];
    r[8 + i]  = (__bf16)b0[i];
    r[12 + i] = (__bf16)b1[i];
  }
  return r;
}

// A (16x32) from bf16 row-major source (used for h in the recurrence).
__device__ __forceinline__ v16bf load_a_bf16(const __bf16* base, size_t lda,
                                             int k0, int lane) {
  const int m = lane & 15, g2 = lane >> 4;
  const __bf16* p = base + (size_t)m * lda + k0 + 8 * g2;
  bf16x8 lo = *(const bf16x8*)(p);
  bf16x8 hi = *(const bf16x8*)(p + 16);
  v16bf r;
#pragma unroll
  for (int i = 0; i < 8; ++i) { r[i] = lo[i]; r[8 + i] = hi[i]; }
  return r;
}

// B (32x16, KxN) from a weight matrix stored [N][K] row-major: a weight row
// IS a B column. Lane n=lane&15 holds K = {8g..8g+7} in halfwords 0..7 and
// K = {8g+16..8g+23} in halfwords 8..15 (g = lane>>4).
__device__ __forceinline__ v16bf load_b_bf16(const __bf16* base, size_t ldb,
                                             int k0, int lane) {
  const int n = lane & 15, g2 = lane >> 4;
  const __bf16* p = base + (size_t)n * ldb + k0 + 8 * g2;
  bf16x8 lo = *(const bf16x8*)(p);
  bf16x8 hi = *(const bf16x8*)(p + 16);
  v16bf r;
#pragma unroll
  for (int i = 0; i < 8; ++i) { r[i] = lo[i]; r[8 + i] = hi[i]; }
  return r;
}

__device__ __forceinline__ v8f wmma_bf16(v16bf a, v16bf b, v8f c) {
  return __builtin_amdgcn_wmma_f32_16x16x32_bf16(false, a, false, b,
                                                 (short)0, c, false, false);
}

__device__ __forceinline__ float sigmf(float x) {
  return 1.0f / (1.0f + __expf(-x));
}

// ---- Kernel 0: f32 -> bf16 weight conversion -------------------------------
__global__ void gru_cvt_bf16(const float* __restrict__ src,
                             __bf16* __restrict__ dst, int n) {
  int i = blockIdx.x * blockDim.x + threadIdx.x;
  if (i < n) dst[i] = (__bf16)src[i];
}

// ---- Kernel 0b: zero h ping-pong buffers + barrier counter -----------------
__global__ void gru_init(float* __restrict__ hf, __bf16* __restrict__ hb,
                         int* __restrict__ bar) {
  int i = blockIdx.x * blockDim.x + threadIdx.x;
  const int n = 2 * B_ * H_;
  if (i < n) { hf[i] = 0.0f; hb[i] = (__bf16)0.0f; }
  if (i == 0) *bar = 0;
}

// ---- Kernel 1: input gates GEMM  IG[t][b][n] = xs @ w_ih^T + b -------------
// grid = (6, 2048), block = 256 (8 waves). Per block: M=64 (batch), N=256.
// Wave tile: 2 m-tiles x 4 n-tiles (square-ish -> minimal fragment traffic,
// 6 independent loads feeding 8 WMMAs per k-step).
__global__ __launch_bounds__(256) void gru_input_gemm(
    const float* __restrict__ xs,    // [B][T][I]
    const __bf16* __restrict__ wih,  // [3H][I] bf16
    const float* __restrict__ bias,  // [3H]
    float* __restrict__ ig) {        // [T][B][3H]
  const int t    = blockIdx.y;
  const int n0   = blockIdx.x * 256;
  const int wave = threadIdx.x >> 5;
  const int lane = threadIdx.x & 31;
  const int mg   = wave & 1;   // m-tiles {2mg, 2mg+1}
  const int ng   = wave >> 1;  // n-tiles {4ng .. 4ng+3}
  const int lm   = lane & 15, lg = lane >> 4;

  const size_t lda = (size_t)T_ * I_;
  const float* a_base0 = xs + (size_t)((2 * mg + 0) * 16) * lda + (size_t)t * I_;
  const float* a_base1 = xs + (size_t)((2 * mg + 1) * 16) * lda + (size_t)t * I_;

  v8f acc[2][4];
#pragma unroll
  for (int mi = 0; mi < 2; ++mi)
#pragma unroll
    for (int j = 0; j < 4; ++j)
      acc[mi][j] = (v8f){0.f, 0.f, 0.f, 0.f, 0.f, 0.f, 0.f, 0.f};

#pragma unroll
  for (int k0 = 0; k0 < I_; k0 += 32) {
    v16bf a0 = load_a_f32(a_base0, lda, k0, lane);
    v16bf a1 = load_a_f32(a_base1, lda, k0, lane);
#pragma unroll
    for (int j = 0; j < 4; ++j) {
      const int n = n0 + (4 * ng + j) * 16;
      v16bf bf = load_b_bf16(wih + (size_t)n * I_, I_, k0, lane);
      acc[0][j] = wmma_bf16(a0, bf, acc[0][j]);
      acc[1][j] = wmma_bf16(a1, bf, acc[1][j]);
    }
  }

  float* igt = ig + (size_t)t * B_ * G3_;
#pragma unroll
  for (int mi = 0; mi < 2; ++mi) {
#pragma unroll
    for (int j = 0; j < 4; ++j) {
      const int n  = n0 + (4 * ng + j) * 16 + lm;
      const float bv = bias[n];
#pragma unroll
      for (int r = 0; r < 8; ++r) {
        const int b = (2 * mg + mi) * 16 + lg * 8 + r;  // C: M = r + 8*(lane>>4)
        igt[(size_t)b * G3_ + n] = acc[mi][j][r] + bv;
      }
    }
  }
}

// ---- Kernel 2: sequential recurrence, 16 cooperating workgroups ------------
// Block g owns h-columns [32g, 32g+32); its 3-gate w_hh slice (96x512 bf16 =
// 96 KB) is DMA'd into LDS by the Tensor Data Mover, then mirrored by a plain
// vector copy (guarantees LDS contents and keeps the stores compiler-visible
// so the ds_load fragment reads are not DCE'd). Resident for all 2048 steps.
__global__ __launch_bounds__(256) void gru_recurrent(
    const float* __restrict__ ig,    // [T][B][3H] f32
    const __bf16* __restrict__ whh,  // [3H][H] bf16
    const float* __restrict__ b_n,   // [H]
    float* __restrict__ h_f32,       // [2][B][H]
    __bf16* __restrict__ h_b16,      // [2][B][H]
    int* __restrict__ bar,
    float* __restrict__ out) {       // [B][H]
  constexpr int G = 16;
  __shared__ __bf16 lds_w[96 * H_];  // ONLY shared object -> LDS byte offset 0
                                     // rows: [gate*32 + local_h][k]
  const int g   = blockIdx.x;
  const int hs0 = g * 32;

#if defined(__gfx1250__) && defined(__HIP_DEVICE_COMPILE__)
  // Stage the three 32x512 bf16 gate segments with TENSOR_LOAD_TO_LDS.
  // D# layout per CDNA5 ISA (2D tile, data_size=2B,
  // tensor_dim0=512, tensor_dim1=32, tile=512x32, dim0_stride=512).
  if (threadIdx.x < 32) {  // single wave issues the DMA; TDM ignores EXEC
#pragma unroll
    for (int seg = 0; seg < 3; ++seg) {
      const unsigned long long ga =
          (unsigned long long)(uintptr_t)(whh + (size_t)(seg * H_ + hs0) * H_);
      const unsigned int lds_off = (unsigned int)(seg * 32 * H_ * 2);
      u32x4 g0;
      g0[0] = 1u;                                   // count=1 valid descriptor
      g0[1] = lds_off;                              // lds_addr
      g0[2] = (unsigned int)(ga & 0xffffffffull);   // global_addr[31:0]
      g0[3] = (unsigned int)((ga >> 32) & 0x1ffffffull) | (2u << 30);  // +type=2
      i32x8 g1;
      g1[0] = (int)(1u << 16);          // data_size=1 (2 bytes), no mask/flags
      g1[1] = (int)(512u << 16);        // tensor_dim0[15:0]=512 at bit 48
      g1[2] = (int)(32u << 16);         // tensor_dim1[15:0]=32 at bit 80
      g1[3] = (int)(512u << 16);        // tile_dim0=512 at bit 112
      g1[4] = 32;                       // tile_dim1=32
      g1[5] = 512;                      // tensor_dim0_stride[31:0]=512
      g1[6] = 0;
      g1[7] = 0;
      i32x4 z4 = (i32x4){0, 0, 0, 0};   // groups 2/3 unused (2D tensor)
#if __clang_major__ >= 23
      i32x8 z8 = (i32x8){0, 0, 0, 0, 0, 0, 0, 0};
      __builtin_amdgcn_tensor_load_to_lds(g0, g1, z4, z4, z8, 0);
#else
      __builtin_amdgcn_tensor_load_to_lds(g0, g1, z4, z4, 0);
#endif
    }
    __builtin_amdgcn_s_wait_tensorcnt(0);
  }
  // Escape the LDS array so the compiler must assume the TDM wrote it.
  asm volatile("" : : "r"(&lds_w[0]) : "memory");
#endif
  __syncthreads();
  // Plain vector copy of the same three 32 KB segments. Writes identical
  // bytes last -> LDS contents are correct regardless of TDM descriptor
  // details, and gives the compiler visible stores to lds_w.
  for (int seg = 0; seg < 3; ++seg) {
    const f32x4* src = (const f32x4*)(whh + (size_t)(seg * H_ + hs0) * H_);
    f32x4* dst = (f32x4*)(lds_w + (size_t)seg * 32 * H_);
    for (int i = threadIdx.x; i < (32 * H_ * 2) / 16; i += 256) dst[i] = src[i];
  }
  __syncthreads();

  const int wave = threadIdx.x >> 5;
  const int lane = threadIdx.x & 31;
  const int mt   = wave & 3;   // batch tile: b = mt*16 ..
  const int hsub = wave >> 2;  // h sub-tile within slice: 0 or 1
  const int lm   = lane & 15, lg = lane >> 4;

  const int   hc   = hs0 + hsub * 16 + lm;  // this lane's h column
  const float bn_v = b_n[hc];

  for (int t = 0; t < T_; ++t) {
    const __bf16* hA = h_b16 + (size_t)(t & 1) * B_ * H_;
    const float* igt = ig + (size_t)t * B_ * G3_;
    const float* hp_buf = h_f32 + (size_t)(t & 1) * B_ * H_;

    // Issue the step-t activation/gate loads FIRST so they overlap the 48
    // WMMAs of the recurrent GEMM (sequential critical path).
    float irv[8], izv[8], inv[8], hpv[8];
#pragma unroll
    for (int r = 0; r < 8; ++r) {
      const int b = mt * 16 + lg * 8 + r;
      const size_t ib = (size_t)b * G3_;
      irv[r] = igt[ib + hc];
      izv[r] = igt[ib + H_ + hc];
      inv[r] = igt[ib + 2 * H_ + hc];
      hpv[r] = hp_buf[(size_t)b * H_ + hc];
    }

    v8f aR = (v8f){0.f,0.f,0.f,0.f,0.f,0.f,0.f,0.f};
    v8f aZ = aR, aN = aR;
#pragma unroll
    for (int k0 = 0; k0 < H_; k0 += 32) {
      v16bf a  = load_a_bf16(hA + (size_t)(mt * 16) * H_, H_, k0, lane);
      v16bf br = load_b_bf16(lds_w + (size_t)(0 * 32 + hsub * 16) * H_, H_, k0, lane);
      v16bf bz = load_b_bf16(lds_w + (size_t)(1 * 32 + hsub * 16) * H_, H_, k0, lane);
      v16bf bn = load_b_bf16(lds_w + (size_t)(2 * 32 + hsub * 16) * H_, H_, k0, lane);
      aR = wmma_bf16(a, br, aR);
      aZ = wmma_bf16(a, bz, aZ);
      aN = wmma_bf16(a, bn, aN);
    }

    float*  hn_buf = h_f32 + (size_t)((t + 1) & 1) * B_ * H_;
    __bf16* hb_buf = h_b16 + (size_t)((t + 1) & 1) * B_ * H_;

    if (t + 1 < T_) {  // warm L2/WGP$ for next step's IG slice
      __builtin_prefetch(ig + (size_t)(t + 1) * B_ * G3_ +
                             (size_t)(mt * 16 + lg * 8) * G3_ + hc, 0, 1);
    }

#pragma unroll
    for (int r = 0; r < 8; ++r) {
      const int b = mt * 16 + lg * 8 + r;
      const float rg = sigmf(irv[r] + aR[r]);
      const float zg = sigmf(izv[r] + aZ[r]);
      const float ng = tanhf(inv[r] + rg * (aN[r] + bn_v));
      const float hn = ng + zg * (hpv[r] - ng);
      const size_t o = (size_t)b * H_ + hc;
      hn_buf[o] = hn;
      hb_buf[o] = (__bf16)hn;
      if (t == T_ - 1) out[o] = hn;
    }

    // device-wide step barrier (monotone counter), plus the CDNA5 cluster
    // barrier (NOP when not cluster-launched per ISA 01_flow_sync).
    __threadfence();
    __syncthreads();
    if (threadIdx.x == 0) {
      atomicAdd(bar, 1);
      const int target = G * (t + 1);
      while (__hip_atomic_load(bar, __ATOMIC_RELAXED,
                               __HIP_MEMORY_SCOPE_AGENT) < target) { }
    }
    __syncthreads();
    __threadfence();
    __builtin_amdgcn_s_cluster_barrier();
  }
}

// ---------------------------------------------------------------------------
extern "C" void kernel_launch(void* const* d_in, const int* in_sizes, int n_in,
                              void* d_out, int out_size, void* d_ws,
                              size_t ws_size, hipStream_t stream) {
  const float* xs   = (const float*)d_in[0];  // [B][T][I]
  const float* w_ih = (const float*)d_in[1];  // [3H][I]
  const float* w_hh = (const float*)d_in[2];  // [3H][H]
  const float* b    = (const float*)d_in[3];  // [3H]
  const float* b_n  = (const float*)d_in[4];  // [H]
  float* out = (float*)d_out;

  char* ws = (char*)d_ws;
  float*  IG     = (float*)(ws + OFF_IG);
  __bf16* wih_bf = (__bf16*)(ws + OFF_WIH);
  __bf16* whh_bf = (__bf16*)(ws + OFF_WHH);
  float*  hf     = (float*)(ws + OFF_HF);
  __bf16* hb     = (__bf16*)(ws + OFF_HB);
  int*    bar    = (int*)(ws + OFF_BAR);

  const int nih = G3_ * I_;
  const int nhh = G3_ * H_;
  gru_cvt_bf16<<<(nih + 255) / 256, 256, 0, stream>>>(w_ih, wih_bf, nih);
  gru_cvt_bf16<<<(nhh + 255) / 256, 256, 0, stream>>>(w_hh, whh_bf, nhh);
  gru_init<<<(2 * B_ * H_ + 255) / 256, 256, 0, stream>>>(hf, hb, bar);

  gru_input_gemm<<<dim3(6, T_), 256, 0, stream>>>(xs, wih_bf, b, IG);
  gru_recurrent<<<16, 256, 0, stream>>>(IG, whh_bf, b_n, hf, hb, bar, out);
}
